// UserHistoryTower_56770877718674
// MI455X (gfx1250) — matrix operand order repaired and checked by reading, not compiled, not run on hardware
//
#include <hip/hip_runtime.h>

typedef __attribute__((ext_vector_type(16))) _Float16 v16h;
typedef __attribute__((ext_vector_type(8)))  _Float16 v8h;
typedef __attribute__((ext_vector_type(8)))  float    v8f;

#define NROWS 8192
#define DIM   128
#define H1DIM 512
#define H2DIM 256

// ---------------------------------------------------------------------------
// Kernel 0: fp32 -> fp16 weight conversion + B-fragment pre-pack.
// Packed layout: Wp[((nt*K_T + kt)*32 + lane)*16 + e]; each lane's 16 halves
// are contiguous (32B) -> load_b is two global_load_b128, coalesced 1KB/wave.
// ---------------------------------------------------------------------------
__global__ void pack_w_kernel(const float* __restrict__ W, _Float16* __restrict__ Wp,
                              int K, int N) {
    const int t = blockIdx.x * blockDim.x + threadIdx.x;
    if (t >= K * N) return;
    const int K_T  = K >> 5;            // k-tiles of 32
    const int e    = t & 15;
    const int lane = (t >> 4) & 31;
    const int frag = t >> 9;
    const int kt   = frag % K_T;
    const int nt   = frag / K_T;
    const int kh   = lane >> 4;
    const int n    = lane & 15;
    Wp[t] = (_Float16)W[(size_t)(kt * 32 + (e >> 3) * 16 + kh * 8 + (e & 7)) * N + nt * 16 + n];
}

// ---------------------------------------------------------------------------
// Kernel 1: sparse masked history aggregation (exact fp32).
// One wave32 per row: predicate -> ballot -> popcount normalizer, then
// cooperative float4 gathers of the rare (~4/row) matching embedding rows.
// ---------------------------------------------------------------------------
__global__ __launch_bounds__(256) void history_kernel(
    const float* __restrict__ A, const int* __restrict__ user,
    const int* __restrict__ ts, const int* __restrict__ click,
    float* __restrict__ hist, float* __restrict__ counts) {
    const int lane = threadIdx.x & 31;
    const int wid  = threadIdx.x >> 5;
    const int row  = blockIdx.x * 8 + wid;

    const int ui = user[row];
    const int ti = ts[row];

    float4 acc = make_float4(0.f, 0.f, 0.f, 0.f);
    int cnt = 0;

    for (int jb = 0; jb < NROWS; jb += 32) {
        __builtin_prefetch(&user[(jb + 1024) & (NROWS - 1)], 0, 0);  // global_prefetch_b8
        __builtin_prefetch(&ts[(jb + 1024) & (NROWS - 1)], 0, 0);

        const int j = jb + lane;
        const bool pred = (click[j] != 0) & (user[j] == ui) & (ti > ts[j]);
        unsigned m = __builtin_amdgcn_ballot_w32(pred);
        cnt += __popc(m);
        while (m) {
            const int t = __builtin_ctz(m);
            m &= m - 1;
            const float4 v = ((const float4*)(A + (size_t)(jb + t) * DIM))[lane];
            acc.x += v.x; acc.y += v.y; acc.z += v.z; acc.w += v.w;
        }
    }

    const float scale = 1.0f / ((float)cnt + 1e-16f);
    float4 r = make_float4(acc.x * scale, acc.y * scale, acc.z * scale, acc.w * scale);
    ((float4*)(hist + (size_t)row * DIM))[lane] = r;
    if (lane == 0) counts[row] = (float)cnt;
}

// ---------------------------------------------------------------------------
// Fragment builders (CDNA5 ISA 7.12.2, wave32). A lane's 16 A-matrix halves
// are two contiguous 8-element runs: K = half*8+[0,8) and 16+half*8+[0,8).
// ---------------------------------------------------------------------------

// A (16x32 f16) from fp32 row-major: 4x b128 loads + packed cvt.
__device__ inline v16h load_a_f32(const float* __restrict__ src, int ld, int kb,
                                  int lane) {
    const int m = lane & 15;
    const int half = lane >> 4;
    const float* p = src + (size_t)m * ld + kb + half * 8;
    const float4 x0 = ((const float4*)p)[0];
    const float4 x1 = ((const float4*)p)[1];
    const float4 x2 = ((const float4*)(p + 16))[0];
    const float4 x3 = ((const float4*)(p + 16))[1];
    v16h a;
    a[0]  = (_Float16)x0.x; a[1]  = (_Float16)x0.y; a[2]  = (_Float16)x0.z; a[3]  = (_Float16)x0.w;
    a[4]  = (_Float16)x1.x; a[5]  = (_Float16)x1.y; a[6]  = (_Float16)x1.z; a[7]  = (_Float16)x1.w;
    a[8]  = (_Float16)x2.x; a[9]  = (_Float16)x2.y; a[10] = (_Float16)x2.z; a[11] = (_Float16)x2.w;
    a[12] = (_Float16)x3.x; a[13] = (_Float16)x3.y; a[14] = (_Float16)x3.z; a[15] = (_Float16)x3.w;
    return a;
}

// A (16x32 f16) from f16 row-major LDS tile: 2x ds_load_b128.
__device__ inline v16h load_a_lds(const _Float16* src, int ld, int kb, int lane) {
    const int m = lane & 15;
    const int half = lane >> 4;
    const _Float16* p = src + m * ld + kb + half * 8;
    const v8h lo = *(const v8h*)p;
    const v8h hi = *(const v8h*)(p + 16);
    return __builtin_shufflevector(lo, hi, 0, 1, 2, 3, 4, 5, 6, 7,
                                   8, 9, 10, 11, 12, 13, 14, 15);
}

// B (32x16 f16) from pre-packed fragments: one 32B contiguous load per lane.
__device__ inline v16h load_b_packed(const _Float16* __restrict__ Wp, int K_T,
                                     int kt, int nt, int lane) {
    return *(const v16h*)(Wp + (((size_t)(nt * K_T + kt) * 32 + lane) << 4));
}

// SiLU with single-instruction hardware reciprocal (v_rcp_f32, ~1 ulp)
// instead of the IEEE v_div_scale/v_fma ladder.
__device__ inline float silu(float x) {
    return x * __builtin_amdgcn_rcpf(1.0f + __expf(-x));
}

// ---------------------------------------------------------------------------
// Kernel 2: 3-layer MLP + double L2 norm. One wave per 16-row tile;
// activations staged in per-wave LDS between layers. Layers 2/3 process two
// n-tiles per iteration so each A-fragment LDS load feeds two WMMAs.
// ---------------------------------------------------------------------------
__global__ __launch_bounds__(64) void mlp_kernel(
    const float* __restrict__ hist, const float* __restrict__ counts,
    const _Float16* __restrict__ W1p, const float* __restrict__ b1,
    const _Float16* __restrict__ W2p, const float* __restrict__ b2,
    const _Float16* __restrict__ W3p, const float* __restrict__ b3,
    float* __restrict__ out) {
    __shared__ _Float16 sh1[2][16 * H1DIM];  // 2 x 16KB (reused as f32 out tile)
    __shared__ _Float16 sh2[2][16 * H2DIM];  // 2 x 8KB
    __shared__ float    sInv[2][16];

    const int lane = threadIdx.x & 31;
    const int wid  = threadIdx.x >> 5;
    const int rowBase = (blockIdx.x * 2 + wid) * 16;

    _Float16* h1 = sh1[wid];
    _Float16* h2 = sh2[wid];
    float* sOut = (float*)sh1[wid];  // h1 region is dead by layer 3

    const int n     = lane & 15;        // D-frag column for this lane
    const int mbase = (lane >> 4) * 8;  // D-frag rows v -> M = mbase+v

    // --- layer 1: [16x128] @ [128x512], A resident in VGPRs ---
    v16h aX[4];
#pragma unroll
    for (int kt = 0; kt < 4; ++kt)
        aX[kt] = load_a_f32(hist + (size_t)rowBase * DIM, DIM, kt * 32, lane);

    for (int nt = 0; nt < H1DIM / 16; nt += 2) {
        v8f c0 = {}, c1 = {};
#pragma unroll
        for (int kt = 0; kt < 4; ++kt) {
            v16h b0 = load_b_packed(W1p, 4, kt, nt, lane);
            v16h b1f = load_b_packed(W1p, 4, kt, nt + 1, lane);
            c0 = __builtin_amdgcn_wmma_f32_16x16x32_f16(false, aX[kt], false, b0,
                                                        (short)0, c0, false, false);
            c1 = __builtin_amdgcn_wmma_f32_16x16x32_f16(false, aX[kt], false, b1f,
                                                        (short)0, c1, false, false);
        }
        const float bias0 = b1[nt * 16 + n];
        const float bias1 = b1[(nt + 1) * 16 + n];
#pragma unroll
        for (int v = 0; v < 8; ++v) {
            h1[(mbase + v) * H1DIM + nt * 16 + n]       = (_Float16)silu(c0[v] + bias0);
            h1[(mbase + v) * H1DIM + (nt + 1) * 16 + n] = (_Float16)silu(c1[v] + bias1);
        }
    }
    __syncthreads();

    // --- layer 2: [16x512] @ [512x256], 2 n-tiles share each A-fragment ---
    for (int nt = 0; nt < H2DIM / 16; nt += 2) {
        v8f c0 = {}, c1 = {};
#pragma unroll
        for (int kt = 0; kt < 16; ++kt) {
            v16h a  = load_a_lds(h1, H1DIM, kt * 32, lane);
            v16h b0 = load_b_packed(W2p, 16, kt, nt, lane);
            v16h b1f = load_b_packed(W2p, 16, kt, nt + 1, lane);
            c0 = __builtin_amdgcn_wmma_f32_16x16x32_f16(false, a, false, b0,
                                                        (short)0, c0, false, false);
            c1 = __builtin_amdgcn_wmma_f32_16x16x32_f16(false, a, false, b1f,
                                                        (short)0, c1, false, false);
        }
        const float bias0 = b2[nt * 16 + n];
        const float bias1 = b2[(nt + 1) * 16 + n];
#pragma unroll
        for (int v = 0; v < 8; ++v) {
            h2[(mbase + v) * H2DIM + nt * 16 + n]       = (_Float16)silu(c0[v] + bias0);
            h2[(mbase + v) * H2DIM + (nt + 1) * 16 + n] = (_Float16)silu(c1[v] + bias1);
        }
    }
    __syncthreads();

    // --- layer 3: [16x256] @ [256x128], fp32 result to LDS ---
    for (int nt = 0; nt < DIM / 16; nt += 2) {
        v8f c0 = {}, c1 = {};
#pragma unroll
        for (int kt = 0; kt < 8; ++kt) {
            v16h a  = load_a_lds(h2, H2DIM, kt * 32, lane);
            v16h b0 = load_b_packed(W3p, 8, kt, nt, lane);
            v16h b1f = load_b_packed(W3p, 8, kt, nt + 1, lane);
            c0 = __builtin_amdgcn_wmma_f32_16x16x32_f16(false, a, false, b0,
                                                        (short)0, c0, false, false);
            c1 = __builtin_amdgcn_wmma_f32_16x16x32_f16(false, a, false, b1f,
                                                        (short)0, c1, false, false);
        }
        const float bias0 = b3[nt * 16 + n];
        const float bias1 = b3[(nt + 1) * 16 + n];
#pragma unroll
        for (int v = 0; v < 8; ++v) {
            sOut[(mbase + v) * DIM + nt * 16 + n]       = c0[v] + bias0;
            sOut[(mbase + v) * DIM + (nt + 1) * 16 + n] = c1[v] + bias1;
        }
    }
    __syncthreads();

    // --- double L2 normalize + empty-row zeroing ---
    {
        const int r  = lane & 15;        // row handled by this lane
        const int hf = lane >> 4;        // column half [hf*64, hf*64+64)
        const float4* rp = (const float4*)(sOut + r * DIM + hf * 64);
        float ss = 0.f;
#pragma unroll
        for (int q = 0; q < 16; ++q) {
            const float4 v = rp[q];
            ss += v.x * v.x + v.y * v.y + v.z * v.z + v.w * v.w;
        }
        ss += __shfl_xor(ss, 16, 32);    // combine the two column halves
        if (lane < 16) {
            const float n1   = sqrtf(ss);
            const float inv1 = 1.0f / fmaxf(n1, 1e-16f);
            const float n2   = n1 * inv1;
            const float inv2 = 1.0f / fmaxf(n2, 1e-16f);
            float f = inv1 * inv2;
            if (counts[rowBase + lane] < 0.5f) f = 0.0f;  // empty history
            sInv[wid][lane] = f;
        }
    }
    __syncthreads();

    {
        float4* out4 = (float4*)(out + (size_t)rowBase * DIM);
        const float4* s4 = (const float4*)sOut;
        for (int idx = lane; idx < 16 * DIM / 4; idx += 32) {
            const int r = idx >> 5;              // 32 float4 per row
            const float f = sInv[wid][r];
            const float4 v = s4[idx];
            out4[idx] = make_float4(v.x * f, v.y * f, v.z * f, v.w * f);
        }
    }
}

// ---------------------------------------------------------------------------
// Host launcher
// ---------------------------------------------------------------------------
extern "C" void kernel_launch(void* const* d_in, const int* in_sizes, int n_in,
                              void* d_out, int out_size, void* d_ws, size_t ws_size,
                              hipStream_t stream) {
    const float* A     = (const float*)d_in[0];
    const int*   user  = (const int*)d_in[1];
    const int*   ts    = (const int*)d_in[2];
    const int*   click = (const int*)d_in[3];
    const float* W1 = (const float*)d_in[4];
    const float* b1 = (const float*)d_in[5];
    const float* W2 = (const float*)d_in[6];
    const float* b2 = (const float*)d_in[7];
    const float* W3 = (const float*)d_in[8];
    const float* b3 = (const float*)d_in[9];
    float* out = (float*)d_out;

    char* ws = (char*)d_ws;
    float*     hist   = (float*)(ws);                 // 4 MB
    float*     counts = (float*)(ws + 4194304);       // 32 KB
    _Float16*  W1p    = (_Float16*)(ws + 4227072);    // 128 KB packed fragments
    _Float16*  W2p    = (_Float16*)(ws + 4358144);    // 256 KB packed fragments
    _Float16*  W3p    = (_Float16*)(ws + 4620288);    // 64 KB packed fragments

    pack_w_kernel<<<(DIM * H1DIM + 255) / 256, 256, 0, stream>>>(W1, W1p, DIM, H1DIM);
    pack_w_kernel<<<(H1DIM * H2DIM + 255) / 256, 256, 0, stream>>>(W2, W2p, H1DIM, H2DIM);
    pack_w_kernel<<<(H2DIM * DIM + 255) / 256, 256, 0, stream>>>(W3, W3p, H2DIM, DIM);

    history_kernel<<<NROWS / 8, 256, 0, stream>>>(A, user, ts, click, hist, counts);

    mlp_kernel<<<NROWS / 32, 64, 0, stream>>>(hist, counts, W1p, b1, W2p, b2,
                                              W3p, b3, out);
}
